// VoxelTransformEncoder_24885040513386
// MI455X (gfx1250) — compile-verified, matrix operand order checked by reading, not run on hardware
//
#include <hip/hip_runtime.h>
#include <hip/hip_bf16.h>
#include <math.h>

// ---------------- problem constants ----------------
#define BATCH 2
#define OWX 130         // x / D dim
#define OHY 34          // y / H dim
#define ODZ 130         // z / W dim
#define NSP (OWX*OHY*ODZ)   // 574600 voxels
#define CF 32
#define PX (OWX+2)
#define PY (OHY+2)
#define PZ (ODZ+2)
#define FH 96
#define FW 96
#define DP 96
#define NBLK_STATS 256
#define ZTILES 9        // ceil(130/16)
#define MT 3            // M-tiles per wave (3*16 = 48 z rows)
#define JOBS_PER_COL 3  // ZTILES / MT
#define GN_EPS 1e-5f

typedef __attribute__((ext_vector_type(16))) _Float16 v16h;
typedef __attribute__((ext_vector_type(8)))  _Float16 v8h;
typedef __attribute__((ext_vector_type(4)))  _Float16 v4h;
typedef __attribute__((ext_vector_type(8)))  float    v8f;

// ---------------- 0) camera / transform matrix ----------------
__device__ __forceinline__ void mat4mul(const float* A, const float* B, float* C) {
  for (int i = 0; i < 4; ++i)
    for (int j = 0; j < 4; ++j) {
      float s = 0.f;
      for (int k = 0; k < 4; ++k) s += A[i*4+k] * B[k*4+j];
      C[i*4+j] = s;
    }
}

__global__ void k_compute_M(const float* azims, const float* elevs,
                            const float* scales, const float* centers, float* Mout) {
  int b = threadIdx.x;
  if (b >= BATCH) return;
  const float d2r = 0.017453292519943295f;
  float az = azims[b] * d2r, el = elevs[b] * d2r;
  float dist = 2000.0f;
  float Cx = dist * cosf(el) * sinf(az);
  float Cy = dist * sinf(el);
  float Cz = dist * cosf(el) * cosf(az);
  float cn = sqrtf(Cx*Cx + Cy*Cy + Cz*Cz);
  float zx = -Cx/cn, zy = -Cy/cn, zz = -Cz/cn;
  // x_ax = cross(up=(0,1,0), z)
  float xx = zz, xy = 0.f, xz = -zx;
  float xn = sqrtf(xx*xx + xy*xy + xz*xz);
  xx /= xn; xy /= xn; xz /= xn;
  // y_ax = cross(z, x)
  float yx = zy*xz - zz*xy, yy = zz*xx - zx*xz, yz = zx*xy - zy*xx;
  float yn = sqrtf(yx*yx + yy*yy + yz*yz);
  yx /= yn; yy /= yn; yz /= yn;
  // R columns = x,y,z axes.  inv(Mw2v) = [[R^T,0],[C,1]]
  float iw[16] = {0};
  iw[0]=xx; iw[1]=xy; iw[2]=xz;
  iw[4]=yx; iw[5]=yy; iw[6]=yz;
  iw[8]=zx; iw[9]=zy; iw[10]=zz;
  iw[12]=Cx; iw[13]=Cy; iw[14]=Cz; iw[15]=1.f;
  // inv(Mproj): s=0.0024, znear=1, zfar=100
  const float s = 0.0024f, zn = 1.0f, zf = 100.0f;
  float ip[16] = {0};
  ip[0] = 1.f/s; ip[5] = 1.f/s; ip[10] = (zf - zn)/s; ip[14] = zn/s; ip[15] = 1.f;
  float munp[16]; mat4mul(ip, iw, munp);
  float mt[16] = {0};
  mt[0] = 1.f/scales[b*3+0]; mt[5] = 1.f/scales[b*3+1]; mt[10] = 1.f/scales[b*3+2];
  mt[12] = centers[b*3+0]; mt[13] = centers[b*3+1]; mt[14] = centers[b*3+2]; mt[15] = 1.f;
  float mc[16] = {0.1f,0,0,0, 0,0.25f,0,0, 0,0,0.1f,0, 65.f,17.f,65.f,1.f};
  float t1[16], M[16];
  mat4mul(munp, mt, t1);
  mat4mul(t1, mc, M);
  for (int i = 0; i < 16; ++i) Mout[b*16+i] = M[i];
}

// ---------------- 1) fill padded f16 act buffer with zeros ----------------
__global__ void k_fill_u32(unsigned int* p, long long n) {
  long long i = (long long)blockIdx.x * blockDim.x + threadIdx.x;
  if (i < n) p[i] = 0u;
}

// ---------------- 2) conv1: 4->32 direct fp32 with analytic coordconv ----------------
__global__ __launch_bounds__(256) void k_conv1(const float* __restrict__ occ,
                                               const float* __restrict__ w1,
                                               const float* __restrict__ b1,
                                               float* __restrict__ y1) {
  __shared__ float sw[CF*4*27 + CF];
  for (int i = threadIdx.x; i < CF*4*27 + CF; i += blockDim.x)
    sw[i] = (i < CF*4*27) ? w1[i] : b1[i - CF*4*27];
  __syncthreads();
  long long t = (long long)blockIdx.x * blockDim.x + threadIdx.x;
  if (t >= (long long)BATCH * NSP) return;
  int z = (int)(t % ODZ);
  int y = (int)((t / ODZ) % OHY);
  int x = (int)((t / ((long long)ODZ*OHY)) % OWX);
  int b = (int)(t / NSP);
  float acc[CF];
#pragma unroll
  for (int oc = 0; oc < CF; ++oc) acc[oc] = sw[CF*4*27 + oc];
  for (int kd = 0; kd < 3; ++kd)
    for (int kh = 0; kh < 3; ++kh)
      for (int kw = 0; kw < 3; ++kw) {
        int xi = x + kd - 1, yi = y + kh - 1, zi = z + kw - 1;
        float i0 = 0.f, i1 = 0.f, i2 = 0.f, i3 = 0.f;
        if (xi >= 0 && xi < OWX && yi >= 0 && yi < OHY && zi >= 0 && zi < ODZ) {
          i0 = occ[((size_t)(b*OWX + xi)*OHY + yi)*ODZ + zi];
          i1 = -1.f + 2.f * xi / (float)(OWX - 1);
          i2 = -1.f + 2.f * yi / (float)(OHY - 1);
          i3 = -1.f + 2.f * zi / (float)(ODZ - 1);
        }
        int tap = (kd*3 + kh)*3 + kw;
#pragma unroll
        for (int oc = 0; oc < CF; ++oc) {
          const float* wv = &sw[(oc*4)*27 + tap];
          acc[oc] += i0*wv[0] + i1*wv[27] + i2*wv[54] + i3*wv[81];
        }
      }
  float* dst = y1 + (size_t)t * CF;
#pragma unroll
  for (int oc = 0; oc < CF; ++oc) dst[oc] = acc[oc];
}

// ---------------- 3) GroupNorm stats (two-stage, deterministic) ----------------
__global__ void k_gn_stage1(const float* __restrict__ src, float* __restrict__ part) {
  __shared__ float red[256];
  int b = blockIdx.y;
  const float* p = src + (size_t)b * NSP * CF;
  float s[8] = {0}, q[8] = {0};
  long long n = (long long)NSP * CF;
  for (long long i = (long long)blockIdx.x * blockDim.x + threadIdx.x; i < n;
       i += (long long)NBLK_STATS * blockDim.x) {
    float v = p[i];
    int g = ((int)(i & 31)) >> 2;
    s[g] += v; q[g] += v * v;
  }
  for (int qi = 0; qi < 16; ++qi) {
    float v = (qi < 8) ? s[qi] : q[qi - 8];
    red[threadIdx.x] = v;
    __syncthreads();
    for (int st = 128; st > 0; st >>= 1) {
      if (threadIdx.x < st) red[threadIdx.x] += red[threadIdx.x + st];
      __syncthreads();
    }
    if (threadIdx.x == 0)
      part[((size_t)b * NBLK_STATS + blockIdx.x) * 16 + qi] = red[0];
    __syncthreads();
  }
}

__global__ void k_gn_stage2(const float* __restrict__ part, float* __restrict__ stats) {
  int t = threadIdx.x;
  if (t >= BATCH * 8) return;
  int b = t >> 3, g = t & 7;
  float s = 0.f, q = 0.f;
  for (int blk = 0; blk < NBLK_STATS; ++blk) {
    const float* pp = &part[((size_t)b * NBLK_STATS + blk) * 16];
    s += pp[g]; q += pp[8 + g];
  }
  float n = 4.0f * (float)NSP;
  float mean = s / n;
  float var = q / n - mean * mean;
  var = var > 0.f ? var : 0.f;
  stats[(b*8 + g)*2 + 0] = mean;
  stats[(b*8 + g)*2 + 1] = rsqrtf(var + GN_EPS);
}

// ---------------- 4) normalize+relu -> padded channels-last f16 ----------------
__global__ void k_norm1(const float* __restrict__ y1, const float* __restrict__ stats,
                        _Float16* __restrict__ act) {
  long long i4 = (long long)blockIdx.x * blockDim.x + threadIdx.x;
  long long n4 = (long long)BATCH * NSP * (CF/4);
  if (i4 >= n4) return;
  long long i = i4 * 4;
  int c = (int)(i & 31);
  int g = c >> 2;
  long long vox = i >> 5;
  int z = (int)(vox % ODZ);
  int y = (int)((vox / ODZ) % OHY);
  int x = (int)((vox / ((long long)ODZ*OHY)) % OWX);
  int b = (int)(vox / NSP);
  const float4 v = *(const float4*)(y1 + i);
  float mean = stats[(b*8 + g)*2 + 0];
  float rstd = stats[(b*8 + g)*2 + 1];
  float r0 = fmaxf((v.x - mean) * rstd, 0.f);
  float r1 = fmaxf((v.y - mean) * rstd, 0.f);
  float r2 = fmaxf((v.z - mean) * rstd, 0.f);
  float r3 = fmaxf((v.w - mean) * rstd, 0.f);
  v4h h; h[0] = (_Float16)r0; h[1] = (_Float16)r1; h[2] = (_Float16)r2; h[3] = (_Float16)r3;
  size_t o = ((((size_t)b*PX + (x+1))*PY + (y+1))*PZ + (z+1))*CF + c;
  *(v4h*)(act + o) = h;
}

// ---------------- 5) pack conv2 weights into CDNA5 B-fragment layout ----------------
// wB[t][nt][lane][j]: lane<16 -> N=lane, K halves {0..7,16..23}; lane>=16 -> K {8..15,24..31}
__global__ void k_prep_wfrag(const float* __restrict__ w2, _Float16* __restrict__ wB) {
  int i = blockIdx.x * blockDim.x + threadIdx.x;
  if (i >= 27 * 2 * 32 * 16) return;
  int j = i & 15;
  int lane = (i >> 4) & 31;
  int nt = (i >> 9) & 1;
  int t = i >> 10;
  int oc = nt * 16 + (lane & 15);
  int ic = ((lane >> 4) & 1) * 8 + (j < 8 ? j : j + 8);
  wB[i] = (_Float16)w2[((size_t)oc * CF + ic) * 27 + t];
}

// ---------------- 6) conv2 implicit GEMM via V_WMMA_F32_16X16X32_F16 ----------------
// One wave owns MT=3 M-tiles (48 z rows) x 2 N-tiles.  All tap strides are
// compile-time constants -> every global/ds load uses an immediate offset.
// Weights staged into LDS with the Tensor Data Mover (TDM) when available.
__global__ __launch_bounds__(256) void k_conv2_wmma(const _Float16* __restrict__ act,
                                                    const _Float16* __restrict__ wB,
                                                    const float* __restrict__ b2,
                                                    float* __restrict__ y2) {
  __shared__ __align__(16) _Float16 sWB[27 * 2 * 32 * 16];

#if defined(__gfx1250__) && __has_builtin(__builtin_amdgcn_tensor_load_to_lds)
  if (threadIdx.x < 32) {
    typedef unsigned int u32x4 __attribute__((ext_vector_type(4)));
    typedef int i32x4 __attribute__((ext_vector_type(4)));
    typedef int i32x8 __attribute__((ext_vector_type(8)));
    typedef __attribute__((address_space(3))) _Float16 lds_f16;
    unsigned ldsAddr = (unsigned)(unsigned long long)(lds_f16*)&sWB[0];
    unsigned long long ga = (unsigned long long)wB;
    const unsigned NU = 27u*2u*32u*16u*2u/8u;      // 6912 8-byte units
    u32x4 g0;
    g0[0] = 1u;                                     // count=1 (valid user D#)
    g0[1] = ldsAddr;                                // lds_addr
    g0[2] = (unsigned)(ga & 0xffffffffu);           // global_addr[31:0]
    g0[3] = (unsigned)((ga >> 32) & 0x1ffffffu) | (2u << 30);  // global_addr[56:32], type=2
    i32x8 g1;
    g1[0] = (int)(3u << 16);                        // data_size = 8 bytes
    g1[1] = (int)((NU & 0xffffu) << 16);            // tensor_dim0[15:0] @ bits63:48
    g1[2] = (int)(((NU >> 16) & 0xffffu) | (1u << 16)); // tensor_dim0[31:16], tensor_dim1=1
    g1[3] = (int)((NU & 0xffffu) << 16);            // tile_dim0 @ bits127:112
    g1[4] = 1;                                      // tile_dim1=1, tile_dim2=0
    g1[5] = (int)NU;                                // tensor_dim0_stride[31:0]
    g1[6] = 0;
    g1[7] = 0;
    i32x4 zz4 = {0, 0, 0, 0};
#if defined(__clang_major__) && (__clang_major__ >= 23)
    i32x8 zz8 = {0, 0, 0, 0, 0, 0, 0, 0};
    __builtin_amdgcn_tensor_load_to_lds(g0, g1, zz4, zz4, zz8, 0);
#else
    __builtin_amdgcn_tensor_load_to_lds(g0, g1, zz4, zz4, 0);
#endif
    __builtin_amdgcn_s_wait_tensorcnt(0);
  }
  __syncthreads();
#else
  {
    const unsigned int* src = (const unsigned int*)wB;
    unsigned int* dst = (unsigned int*)sWB;
    for (int i = threadIdx.x; i < 27 * 2 * 32 * 16 / 2; i += blockDim.x) dst[i] = src[i];
  }
  __syncthreads();
#endif

  int wave = threadIdx.x >> 5;
  int lane = threadIdx.x & 31;
  long long wid = (long long)blockIdx.x * 8 + wave;
  const long long NJOB = (long long)BATCH * OWX * OHY * JOBS_PER_COL;
  if (wid >= NJOB) return;               // wave-uniform exit (EXEC stays all-ones)
  int jt = (int)(wid % JOBS_PER_COL);
  long long r = wid / JOBS_PER_COL;
  int y = (int)(r % OHY); r /= OHY;
  int x = (int)(r % OWX);
  int b = (int)(r / OWX);
  int mrow = lane & 15;
  int hi = lane >> 4;
  int cb = hi * 8;
  float bias0 = b2[lane & 15];
  float bias1 = b2[16 + (lane & 15)];

  // per-tile base pointers (z clamped; clamped rows discarded at store)
  const size_t rowbase = (((size_t)b*PX + (x+1))*PY + (y+1))*PZ;
  const _Float16* base[MT];
#pragma unroll
  for (int tl = 0; tl < MT; ++tl) {
    int z = (jt*MT + tl)*16 + mrow;
    int zc = z < ODZ ? z : ODZ - 1;
    base[tl] = act + (rowbase + zc + 1)*CF + cb;
  }
  const _Float16* wlane = &sWB[lane * 16];

  v8f acc[MT][2];
#pragma unroll
  for (int tl = 0; tl < MT; ++tl) { acc[tl][0] = (v8f){}; acc[tl][1] = (v8f){}; }

#pragma unroll
  for (int t = 0; t < 27; ++t) {
    const int dx = t/9 - 1, dy = (t/3)%3 - 1, dz = t%3 - 1;
    const int delta = ((dx*PY + dy)*PZ + dz)*CF;      // compile-time constant (halfs)
    const int woff  = t * 1024;                       // compile-time constant (halfs)
    v8h b0l = *(const v8h*)(wlane + woff);
    v8h b0h = *(const v8h*)(wlane + woff + 8);
    v8h b1l = *(const v8h*)(wlane + woff + 512);
    v8h b1h = *(const v8h*)(wlane + woff + 520);
    v16h bb0 = __builtin_shufflevector(b0l, b0h, 0,1,2,3,4,5,6,7,8,9,10,11,12,13,14,15);
    v16h bb1 = __builtin_shufflevector(b1l, b1h, 0,1,2,3,4,5,6,7,8,9,10,11,12,13,14,15);
#pragma unroll
    for (int tl = 0; tl < MT; ++tl) {
      v8h al = *(const v8h*)(base[tl] + delta);
      v8h ah = *(const v8h*)(base[tl] + delta + 16);
      v16h a = __builtin_shufflevector(al, ah, 0,1,2,3,4,5,6,7,8,9,10,11,12,13,14,15);
      acc[tl][0] = __builtin_amdgcn_wmma_f32_16x16x32_f16(false, a, false, bb0, (short)0, acc[tl][0], false, false);
      acc[tl][1] = __builtin_amdgcn_wmma_f32_16x16x32_f16(false, a, false, bb1, (short)0, acc[tl][1], false, false);
    }
  }

#pragma unroll
  for (int tl = 0; tl < MT; ++tl) {
    int ztile = (jt*MT + tl) * 16;
#pragma unroll
    for (int rr = 0; rr < 8; ++rr) {
      int m = ztile + rr + hi * 8;       // C layout: lanes 0-15 -> M=rr, 16-31 -> M=rr+8
      if (m < ODZ) {
        size_t o = (((size_t)(b*OWX + x)*OHY + y)*ODZ + m)*CF + (lane & 15);
        y2[o]      = acc[tl][0][rr] + bias0;
        y2[o + 16] = acc[tl][1][rr] + bias1;
      }
    }
  }
}

// ---------------- 7) normalize+relu in place (fp32) ----------------
__global__ void k_norm2(float* __restrict__ y2, const float* __restrict__ stats) {
  long long i4 = (long long)blockIdx.x * blockDim.x + threadIdx.x;
  long long n4 = (long long)BATCH * NSP * (CF/4);
  if (i4 >= n4) return;
  long long i = i4 * 4;
  int c = (int)(i & 31);
  int g = c >> 2;
  int b = (int)(i / ((long long)NSP * CF));
  float mean = stats[(b*8 + g)*2 + 0];
  float rstd = stats[(b*8 + g)*2 + 1];
  float4 v = *(const float4*)(y2 + i);
  v.x = fmaxf((v.x - mean) * rstd, 0.f);
  v.y = fmaxf((v.y - mean) * rstd, 0.f);
  v.z = fmaxf((v.z - mean) * rstd, 0.f);
  v.w = fmaxf((v.w - mean) * rstd, 0.f);
  *(float4*)(y2 + i) = v;
}

// ---------------- 8) ray-march + first-hit feature sampling ----------------
__device__ __forceinline__ float tri_occ(const float* vol, float u, float v, float w) {
  float fu = floorf(u), fv = floorf(v), fw = floorf(w);
  float du = u - fu, dv = v - fv, dw = w - fw;
  int x0 = (int)fu, y0 = (int)fv, z0 = (int)fw;
  float acc = 0.f;
#pragma unroll
  for (int c = 0; c < 8; ++c) {
    int dx = c & 1, dy = (c >> 1) & 1, dz = c >> 2;
    int xi = x0 + dx, yi = y0 + dy, zi = z0 + dz;
    if (xi < 0 || xi >= OWX || yi < 0 || yi >= OHY || zi < 0 || zi >= ODZ) continue;
    float wt = (dx ? du : 1.f - du) * (dy ? dv : 1.f - dv) * (dz ? dw : 1.f - dw);
    acc += wt * vol[((size_t)xi * OHY + yi) * ODZ + zi];
  }
  return acc;
}

__global__ __launch_bounds__(256) void k_sample(const float* __restrict__ occ,
                                                const float* __restrict__ feat,
                                                const float* __restrict__ Mm,
                                                float* __restrict__ out) {
  int wave = threadIdx.x >> 5;
  int lane = threadIdx.x & 31;
  int pid = blockIdx.x * (blockDim.x >> 5) + wave;
  if (pid >= BATCH * FW * FH) return;
  int i = pid % FH;
  int j = (pid / FH) % FW;
  int b = pid / (FH * FW);
  const float* M = Mm + b * 16;
  float gx = 1.f - 2.f * i / (float)(FH - 1);
  float gy = 1.f - 2.f * j / (float)(FW - 1);
  float bx = gx*M[0] + gy*M[4] + M[12];
  float by = gx*M[1] + gy*M[5] + M[13];
  float bz = gx*M[2] + gy*M[6] + M[14];
  float drx = M[8], dry = M[9], drz = M[10];
  const float su = (float)(OWX - 1) / (float)OWX;
  const float sv = (float)(OHY - 1) / (float)OHY;
  const float sw = (float)(ODZ - 1) / (float)ODZ;
  const float zstep = 0.03f / (float)(DP - 1);
  const float* ov = occ + (size_t)b * NSP;

  int dstar = 0;
  bool found = false;
  for (int k = 0; k < 3 && !found; ++k) {
    int d = k * 32 + lane;
    float zd = 0.03f + d * zstep;
    float u = (bx + zd * drx) * su;
    float v = (by + zd * dry) * sv;
    float w = (bz + zd * drz) * sw;
    float o = tri_occ(ov, u, v, w);
    unsigned long long m64 = __ballot(o > 0.1f);
    unsigned mask = (unsigned)m64;
    if (mask) { dstar = k * 32 + __builtin_ctz(mask); found = true; }
  }
  // sample 32 features at depth dstar; lane == channel
  float zd = 0.03f + dstar * zstep;
  float u = (bx + zd * drx) * su;
  float v = (by + zd * dry) * sv;
  float w = (bz + zd * drz) * sw;
  float fu = floorf(u), fv = floorf(v), fw = floorf(w);
  float du = u - fu, dv = v - fv, dw = w - fw;
  int x0 = (int)fu, y0 = (int)fv, z0 = (int)fw;
  float acc = 0.f;
#pragma unroll
  for (int c = 0; c < 8; ++c) {
    int dx = c & 1, dy = (c >> 1) & 1, dz = c >> 2;
    int xi = x0 + dx, yi = y0 + dy, zi = z0 + dz;
    if (xi < 0 || xi >= OWX || yi < 0 || yi >= OHY || zi < 0 || zi >= ODZ) continue;
    float wt = (dx ? du : 1.f - du) * (dy ? dv : 1.f - dv) * (dz ? dw : 1.f - dw);
    acc += wt * feat[((((size_t)b*OWX + xi)*OHY + yi)*ODZ + zi)*CF + lane];
  }
  out[(((size_t)b*CF + lane)*FW + j)*FH + i] = acc;
}

// ---------------- launch ----------------
extern "C" void kernel_launch(void* const* d_in, const int* in_sizes, int n_in,
                              void* d_out, int out_size, void* d_ws, size_t ws_size,
                              hipStream_t stream) {
  const float* occ     = (const float*)d_in[0];
  const float* azims   = (const float*)d_in[1];
  const float* elevs   = (const float*)d_in[2];
  const float* scales  = (const float*)d_in[3];
  const float* centers = (const float*)d_in[4];
  const float* w1      = (const float*)d_in[5];
  const float* b1      = (const float*)d_in[6];
  const float* w2      = (const float*)d_in[7];
  const float* b2      = (const float*)d_in[8];
  float* out = (float*)d_out;

  char* ws = (char*)d_ws;
  float*    Mws   = (float*)(ws + 0);                        // 128 B
  float*    stats = (float*)(ws + 256);                      // 128 B
  float*    part  = (float*)(ws + 1024);                     // 32 KB
  _Float16* wB    = (_Float16*)(ws + 64 * 1024);             // 55 KB
  float*    regA  = (float*)(ws + (size_t)(1 << 20));        // 147 MB (y1 -> y2 -> features)
  _Float16* actB  = (_Float16*)(ws + (size_t)(1 << 20) + (size_t)160 * 1024 * 1024); // 80 MB

  // 0) transform matrices
  k_compute_M<<<1, 32, 0, stream>>>(azims, elevs, scales, centers, Mws);

  // 1) zero padded act buffer
  long long padU32 = (long long)BATCH * PX * PY * PZ * CF / 2;
  k_fill_u32<<<(int)((padU32 + 255) / 256), 256, 0, stream>>>((unsigned int*)actB, padU32);

  // 2) conv1
  long long nvox = (long long)BATCH * NSP;
  k_conv1<<<(int)((nvox + 255) / 256), 256, 0, stream>>>(occ, w1, b1, regA);

  // 3) GN1 stats + normalize -> padded f16
  k_gn_stage1<<<dim3(NBLK_STATS, BATCH), 256, 0, stream>>>(regA, part);
  k_gn_stage2<<<1, 64, 0, stream>>>(part, stats);
  long long n4 = (long long)BATCH * NSP * (CF / 4);
  k_norm1<<<(int)((n4 + 255) / 256), 256, 0, stream>>>(regA, stats, actB);

  // 4) weight fragments + conv2 WMMA
  k_prep_wfrag<<<(27 * 2 * 32 * 16 + 255) / 256, 256, 0, stream>>>(w2, wB);
  long long njobs = (long long)BATCH * OWX * OHY * JOBS_PER_COL;
  k_conv2_wmma<<<(int)((njobs + 7) / 8), 256, 0, stream>>>(actB, wB, b2, regA);

  // 5) GN2 stats + in-place normalize
  k_gn_stage1<<<dim3(NBLK_STATS, BATCH), 256, 0, stream>>>(regA, part);
  k_gn_stage2<<<1, 64, 0, stream>>>(part, stats);
  k_norm2<<<(int)((n4 + 255) / 256), 256, 0, stream>>>(regA, stats);

  // 6) ray-march + first-hit sampling
  int npix = BATCH * FW * FH;
  k_sample<<<(npix + 7) / 8, 256, 0, stream>>>(occ, regA, Mws, out);
}
